// LeNet_335007449253
// MI455X (gfx1250) — matrix operand order repaired and checked by reading, not compiled, not run on hardware
//
#include <hip/hip_runtime.h>
#include <hip/hip_bf16.h>
#include <stddef.h>
#include <stdint.h>

// ---------------------------------------------------------------------------
// Types for CDNA5 WMMA (wave32): D(16x16 f32) = A(16x32 f16) * B(32x16 f16) + C
// ---------------------------------------------------------------------------
typedef __attribute__((ext_vector_type(16))) _Float16 v16h;
typedef __attribute__((ext_vector_type(8)))  _Float16 v8h;
typedef __attribute__((ext_vector_type(8)))  float    v8f;
typedef __attribute__((ext_vector_type(4)))  float    v4f;
// TDM descriptor group types (6-arg builtin form on this toolchain)
typedef __attribute__((ext_vector_type(4))) unsigned int u32x4;
typedef __attribute__((ext_vector_type(8))) int          i32x8;
typedef __attribute__((ext_vector_type(4))) int          i32x4;

__device__ __forceinline__ v8f wmma_f16(v16h a, v16h b, v8f c) {
    return __builtin_amdgcn_wmma_f32_16x16x32_f16(
        /*neg_a=*/false, a, /*neg_b=*/false, b,
        /*c_mod=*/(short)0, c, /*reuse_a=*/false, /*reuse_b=*/false);
}

__device__ __forceinline__ float sigmoidf_(float x) {
    return 1.0f / (1.0f + __expf(-x));
}

// A-fragment loader (16-bit A 16x32, contiguous k order used here, same as packed B):
//   lanes 0-15 : M=lane,    e0..15 -> K = kb+e
//   lanes 16-31: M=lane-16, e0..15 -> K = kb+16+e
__device__ __forceinline__ v16h load_A_rowmajor(const _Float16* __restrict__ base,
                                                int ldk, int kbase, int lane) {
    const int m   = lane & 15;
    const int kh16 = (lane & 16) ? 16 : 0;
    const _Float16* p = base + m * ldk + kbase + kh16;
    v16h a;
#pragma unroll
    for (int i = 0; i < 8; ++i) a[i] = p[i];
#pragma unroll
    for (int i = 0; i < 8; ++i) a[8 + i] = p[8 + i];
    return a;
}

// ---------------------------------------------------------------------------
// Problem constants
// ---------------------------------------------------------------------------
#define BATCH   16
#define HW      512
#define PLANE   (512 * 512)           // 262144
#define C1_OUT  6
#define C1_H    508
#define C1_P    (508 * 508)           // 258064 = 16 * 16129
#define P1_H    254
#define C2_OUT  16
#define C2_H    250
#define C2_P    (250 * 250)           // 62500 -> 3907 tiles of 16 (last ragged)
#define P2_H    125
#define FC0_K   250000
#define FC0_N   120
#define FC1_N   84

// Workspace layout (bytes)
static constexpr size_t OFF_BUFA   = 0;                         // 33,554,432  x4 / pooled1 / pooled2
static constexpr size_t OFF_BUFB   = 33554432;                  // 49,548,288  conv1out / conv2out
static constexpr size_t OFF_MASK   = OFF_BUFB + 49548288;       //  4,194,304  bright mask bytes
static constexpr size_t OFF_WP1    = OFF_MASK + 4194304;        //      4,096  packed conv1 B frags
static constexpr size_t OFF_WP2    = OFF_WP1 + 4096;            //      5,120  packed conv2 B frags
static constexpr size_t OFF_WPFC1  = OFF_WP2 + 5120;            //     24,576  packed fc1 B frags
static constexpr size_t OFF_ACC    = OFF_WPFC1 + 24576;         //      8,192  fc0 f32 accumulator 16x128
static constexpr size_t OFF_H0     = OFF_ACC + 8192;            //      4,096  h0 f16 16x128 (padded)
static constexpr size_t OFF_H1     = OFF_H0 + 4096;             //      5,632  h1 f32 16x84
static constexpr size_t OFF_PARAMS = OFF_H1 + 5632;             //        192  (g1,g2,gain) x 16

// ---------------------------------------------------------------------------
// Pack weights (row-major W[N][K], f32) into WMMA B-register layout, f16.
// B order used here (contiguous k, as in the A loader): lane holds column n = lane&15;
//   lanes 0-15: element e -> K = kb+e; lanes 16-31: element e -> K = kb+16+e.
// Output: [ntile][kstep][lane][e] halves -> one v16h load per lane per kstep.
// ---------------------------------------------------------------------------
__global__ void pack_B_kernel(const float* __restrict__ W, _Float16* __restrict__ out,
                              int N, int K, int nks, int ntiles) {
    int idx = blockIdx.x * blockDim.x + threadIdx.x;
    int total = ntiles * nks * 32 * 16;
    if (idx >= total) return;
    int e    = idx & 15;
    int lane = (idx >> 4) & 31;
    int ks   = (idx >> 9) % nks;
    int nt   = idx / (nks * 512);
    int k    = ks * 32 + ((lane & 16) ? 16 : 0) + e;
    int n    = nt * 16 + (lane & 15);
    float v  = (k < K && n < N) ? W[(size_t)n * K + k] : 0.0f;
    out[idx] = (_Float16)v;
}

// ---------------------------------------------------------------------------
// Prep: grayscale mean, bright mask, f16 4-channel input tensor (img ++ I)
// ---------------------------------------------------------------------------
__global__ void prep_kernel(const float* __restrict__ img, _Float16* __restrict__ x4,
                            unsigned char* __restrict__ mask) {
    int idx = blockIdx.x * blockDim.x + threadIdx.x;   // BATCH*PLANE threads
    if (idx >= BATCH * PLANE) return;
    int b   = idx >> 18;
    int pix = idx & (PLANE - 1);
    float r  = img[(size_t)(b * 3 + 0) * PLANE + pix];
    float g  = img[(size_t)(b * 3 + 1) * PLANE + pix];
    float bl = img[(size_t)(b * 3 + 2) * PLANE + pix];
    float I  = (r + g + bl) * (1.0f / 3.0f);
    x4[(size_t)(b * 4 + 0) * PLANE + pix] = (_Float16)r;
    x4[(size_t)(b * 4 + 1) * PLANE + pix] = (_Float16)g;
    x4[(size_t)(b * 4 + 2) * PLANE + pix] = (_Float16)bl;
    x4[(size_t)(b * 4 + 3) * PLANE + pix] = (_Float16)I;
    mask[idx] = (I > 0.5f) ? 1 : 0;
}

// ---------------------------------------------------------------------------
// conv1: implicit GEMM, one wave per 16-pixel tile (C1_P divisible by 16).
// K = 4*25 = 100 padded to 128 (4 wmma k-steps). N = 6 padded to 16.
// Staging: lane owns patch row m = lane&15; two lanes split the 20 (ic,kr)
// kernel rows; each copy is 5 contiguous halves -> one div per lane total.
// ---------------------------------------------------------------------------
#define C1_LDKP 136   // 128 + 8 halves pad to break LDS bank conflicts
__global__ __launch_bounds__(256) void conv1_kernel(const _Float16* __restrict__ x4,
                                                    const _Float16* __restrict__ wp,
                                                    const float* __restrict__ bias,
                                                    _Float16* __restrict__ out) {
    __shared__ _Float16 lds[8 * 16 * C1_LDKP];
    const int tid  = threadIdx.x;
    const int wave = tid >> 5;
    const int lane = tid & 31;
    const int tile = blockIdx.x * 8 + wave;            // < 16 * 16129
    const int b    = tile / 16129;
    const int pp   = tile - b * 16129;
    _Float16* tbase = lds + wave * (16 * C1_LDKP);

    const int m    = lane & 15;
    const int half = lane >> 4;
    {
        const int p  = pp * 16 + m;
        const int oy = p / C1_H;
        const int ox = p - oy * C1_H;
        _Float16* drow = tbase + m * C1_LDKP;
        const _Float16* s0 = x4 + (size_t)b * 4 * PLANE + (size_t)oy * HW + ox;
#pragma unroll
        for (int ic = 0; ic < 4; ++ic) {
#pragma unroll
            for (int kr = 0; kr < 5; ++kr) {
                if (((ic * 5 + kr) & 1) == half) {
                    const _Float16* s = s0 + (size_t)ic * PLANE + kr * HW;
                    _Float16* d = drow + ic * 25 + kr * 5;
#pragma unroll
                    for (int kc = 0; kc < 5; ++kc) d[kc] = s[kc];
                }
            }
        }
#pragma unroll
        for (int i = 0; i < 14; ++i)   // zero pad K = 100..127
            drow[100 + half * 14 + i] = (_Float16)0.0f;
    }
    __syncthreads();

    v8f c = {};
#pragma unroll
    for (int ks = 0; ks < 4; ++ks) {
        v16h a  = load_A_rowmajor(tbase, C1_LDKP, ks * 32, lane);
        v16h bb = *(const v16h*)(wp + ((ks * 32 + lane) << 4));
        c = wmma_f16(a, bb, c);
    }

    const int n  = lane & 15;
    const int mb = (lane & 16) ? 8 : 0;
    if (n < C1_OUT) {
        const int p0  = pp * 16;
        const int oy0 = p0 / C1_H;
        const int ox0 = p0 - oy0 * C1_H;
#pragma unroll
        for (int r = 0; r < 8; ++r) {
            int d  = r + mb;
            int ox = ox0 + d, oy = oy0;
            if (ox >= C1_H) { ox -= C1_H; oy += 1; }   // single wrap, d < 16
            float v = sigmoidf_(c[r] + bias[n]);
            out[(size_t)(((b * C1_OUT + n) * C1_H + oy) * C1_H) + ox] = (_Float16)v;
        }
    }
}

// ---------------------------------------------------------------------------
// conv2: implicit GEMM. K = 6*25 = 150 padded to 160 (5 k-steps). N = 16.
// 62500 pixels -> 3907 tiles per batch, last tile ragged (p < 62500 mask).
// ---------------------------------------------------------------------------
#define C2_LDKP 168
__global__ __launch_bounds__(256) void conv2_kernel(const _Float16* __restrict__ xin,
                                                    const _Float16* __restrict__ wp,
                                                    const float* __restrict__ bias,
                                                    _Float16* __restrict__ out) {
    __shared__ _Float16 lds[8 * 16 * C2_LDKP];
    const int tid  = threadIdx.x;
    const int wave = tid >> 5;
    const int lane = tid & 31;
    const int tile = blockIdx.x * 8 + wave;            // < 16 * 3907
    const int b    = tile / 3907;
    const int pp   = tile - b * 3907;
    _Float16* tbase = lds + wave * (16 * C2_LDKP);

    const int m    = lane & 15;
    const int half = lane >> 4;
    {
        const int p     = pp * 16 + m;
        const bool good = (p < C2_P);
        const int pc    = good ? p : 0;
        const int oy    = pc / C2_H;
        const int ox    = pc - oy * C2_H;
        _Float16* drow  = tbase + m * C2_LDKP;
        const _Float16* s0 = xin + (size_t)b * C1_OUT * (P1_H * P1_H)
                                 + (size_t)oy * P1_H + ox;
#pragma unroll
        for (int ic = 0; ic < 6; ++ic) {
#pragma unroll
            for (int kr = 0; kr < 5; ++kr) {
                if (((ic * 5 + kr) & 1) == half) {
                    const _Float16* s = s0 + (size_t)ic * (P1_H * P1_H) + kr * P1_H;
                    _Float16* d = drow + ic * 25 + kr * 5;
#pragma unroll
                    for (int kc = 0; kc < 5; ++kc)
                        d[kc] = good ? s[kc] : (_Float16)0.0f;
                }
            }
        }
#pragma unroll
        for (int i = 0; i < 5; ++i)    // zero pad K = 150..159
            drow[150 + half * 5 + i] = (_Float16)0.0f;
    }
    __syncthreads();

    v8f c = {};
#pragma unroll
    for (int ks = 0; ks < 5; ++ks) {
        v16h a  = load_A_rowmajor(tbase, C2_LDKP, ks * 32, lane);
        v16h bb = *(const v16h*)(wp + ((ks * 32 + lane) << 4));
        c = wmma_f16(a, bb, c);
    }

    const int n  = lane & 15;
    const int mb = (lane & 16) ? 8 : 0;
    const int p0  = pp * 16;
    const int oy0 = p0 / C2_H;
    const int ox0 = p0 - oy0 * C2_H;
#pragma unroll
    for (int r = 0; r < 8; ++r) {
        int d = r + mb;
        if (p0 + d < C2_P) {
            int ox = ox0 + d, oy = oy0;
            if (ox >= C2_H) { ox -= C2_H; oy += 1; }
            float v = sigmoidf_(c[r] + bias[n]);
            out[(size_t)(((b * C2_OUT + n) * C2_H + oy) * C2_H) + ox] = (_Float16)v;
        }
    }
}

// ---------------------------------------------------------------------------
// 2x2 max pool, f16 -> f16
// ---------------------------------------------------------------------------
__global__ void pool_kernel(const _Float16* __restrict__ in, _Float16* __restrict__ out,
                            int C, int Hin, int Hout, int total) {
    int idx = blockIdx.x * blockDim.x + threadIdx.x;
    if (idx >= total) return;
    int x = idx % Hout;
    int t = idx / Hout;
    int y = t % Hout; t /= Hout;
    int c = t % C;
    int b = t / C;
    const _Float16* p = in + (size_t)(((b * C + c) * Hin + 2 * y) * Hin) + 2 * x;
    float a0 = (float)p[0],   a1 = (float)p[1];
    float a2 = (float)p[Hin], a3 = (float)p[Hin + 1];
    out[idx] = (_Float16)fmaxf(fmaxf(a0, a1), fmaxf(a2, a3));
}

__global__ void zero_kernel(float* __restrict__ p, int n) {
    int i = blockIdx.x * blockDim.x + threadIdx.x;
    if (i < n) p[i] = 0.0f;
}

// ---------------------------------------------------------------------------
// fc0: (16 x 250000) @ (250000 x 120) via split-K WMMA + f32 atomics.
// Grid: 256 blocks of 1 wave: block = ntile(0..7)*32 + ksplit(0..31).
// A rows (batch) are contiguous f16; B converted f32->f16 on load, all
// operand fetches explicit 128-bit vector loads (bases 16B aligned).
// ---------------------------------------------------------------------------
__global__ __launch_bounds__(32) void fc0_kernel(const _Float16* __restrict__ X,
                                                 const float* __restrict__ W,
                                                 float* __restrict__ acc) {
    const int lane  = threadIdx.x & 31;
    const int nt    = blockIdx.x >> 5;          // 0..7
    const int split = blockIdx.x & 31;          // 0..31
    const int n     = nt * 16 + (lane & 15);
    const int m     = lane & 15;                // batch row
    const int kh8   = (lane & 16) ? 8 : 0;
    const int kh16  = (lane & 16) ? 16 : 0;
    const int nfull = FC0_K / 32;               // 7812 full k-steps
    const int ks0   = (split * nfull) >> 5;
    const int ks1   = ((split + 1) * nfull) >> 5;

    const _Float16* Xrow = X + (size_t)m * FC0_K;
    const float*    Wrow = W + (size_t)((n < FC0_N) ? n : 0) * FC0_K;

    v8f c = {};
    for (int ks = ks0; ks < ks1; ++ks) {
        const int kb = ks * 32;
        __builtin_prefetch(Wrow + kb + 512, 0, 0);   // global_prefetch_b8, next weight chunk
        v8h a0 = *(const v8h*)(Xrow + kb + kh16);
        v8h a1 = *(const v8h*)(Xrow + kb + kh16 + 8);
        const v4f* pw = (const v4f*)(Wrow + kb + kh16);
        v4f w0 = pw[0], w1 = pw[1], w2 = pw[2], w3 = pw[3];
        v16h a, bb;
#pragma unroll
        for (int i = 0; i < 8; ++i) { a[i] = a0[i]; a[8 + i] = a1[i]; }
#pragma unroll
        for (int i = 0; i < 4; ++i) {
            bb[i]      = (_Float16)w0[i];
            bb[4 + i]  = (_Float16)w1[i];
            bb[8 + i]  = (_Float16)w2[i];
            bb[12 + i] = (_Float16)w3[i];
        }
        c = wmma_f16(a, bb, c);
    }

    if (split == 31) {   // ragged tail: K = 249984..249999 valid (uniform branch)
        const int kb = nfull * 32;
        v16h a, bb;
#pragma unroll
        for (int i = 0; i < 8; ++i) {
            int k = kb + kh16 + i;
            a[i] = (k < FC0_K) ? Xrow[k] : (_Float16)0.0f;
        }
#pragma unroll
        for (int i = 0; i < 8; ++i) {
            int k = kb + kh16 + 8 + i;
            a[8 + i] = (k < FC0_K) ? Xrow[k] : (_Float16)0.0f;
        }
#pragma unroll
        for (int i = 0; i < 16; ++i) {
            int k = kb + kh16 + i;
            bb[i] = (k < FC0_K) ? (_Float16)Wrow[k] : (_Float16)0.0f;
        }
        c = wmma_f16(a, bb, c);
    }

#pragma unroll
    for (int r = 0; r < 8; ++r) {
        int row = r + ((lane & 16) ? 8 : 0);
        atomicAdd(&acc[row * 128 + nt * 16 + (lane & 15)], c[r]);
    }
}

__global__ void fc0_post_kernel(const float* __restrict__ acc, const float* __restrict__ bias,
                                _Float16* __restrict__ h0) {
    int idx = blockIdx.x * blockDim.x + threadIdx.x;   // 16*128
    if (idx >= 16 * 128) return;
    int n = idx & 127;
    float v = 0.0f;
    if (n < FC0_N) v = sigmoidf_(acc[idx] + bias[n]);
    h0[idx] = (_Float16)v;   // columns >=120 zero-padded for fc1 A staging
}

// ---------------------------------------------------------------------------
// fc1: (16 x 120) @ (120 x 84) -> 6 N-tiles, K padded to 128 (4 k-steps).
// One block, 6 waves. h0 (4KB contiguous) staged into padded LDS rows via the
// Tensor Data Mover when available: 1-D tile of 2048 f16 elements with LDS
// padding of 4 dwords after every 64 dwords (-> 136-half rows), fenced with
// s_wait_tensorcnt. Fallback: cooperative per-element copy.
// ---------------------------------------------------------------------------
__global__ __launch_bounds__(192) void fc1_kernel(const _Float16* __restrict__ h0,
                                                  const _Float16* __restrict__ wp,
                                                  const float* __restrict__ bias,
                                                  float* __restrict__ h1) {
    __shared__ _Float16 lds[16 * C1_LDKP];
    const int tid  = threadIdx.x;
    const int wave = tid >> 5;
    const int lane = tid & 31;

#if __has_builtin(__builtin_amdgcn_tensor_load_to_lds) && __has_builtin(__builtin_amdgcn_s_wait_tensorcnt)
    if (wave == 0) {
        // D# group 0: count=1, type=2(image), lds_addr, 57-bit global_addr
        uint64_t ga = (uint64_t)(uintptr_t)h0;
        unsigned  la = (unsigned)(uintptr_t)(void*)lds;   // LDS byte offset
        u32x4 g0;
        g0[0] = 1u;                                       // count=1, user mode
        g0[1] = la;                                       // lds_addr
        g0[2] = (unsigned)(ga & 0xFFFFFFFFu);             // global_addr[31:0]
        g0[3] = (unsigned)((ga >> 32) & 0x01FFFFFFu) | (2u << 30);  // addr[56:32] | type=2
        // D# group 1: data_size=2B(enc 1), pad_enable, pad_interval=64dw(enc 5),
        // pad_amount=4dw(enc 3); tensor_dim0=tile_dim0=2048, stride0=2048.
        i32x8 g1;
        g1[0] = (1 << 16) | (1 << 20) | (5 << 22) | (3 << 25);
        g1[1] = (2048 & 0xFFFF) << 16;                    // tensor_dim0 lo16 in [31:16]
        g1[2] = (2048 >> 16) | (1 << 16);                 // tensor_dim0 hi16 | tensor_dim1=1
        g1[3] = (2048 << 16);                             // tile_dim0 in [31:16]
        g1[4] = 0;                                        // tile_dim1=0 (unused), tile_dim2=0
        g1[5] = 2048;                                     // tensor_dim0_stride lo32
        g1[6] = 0;
        g1[7] = 0;
        i32x4 gz  = {0, 0, 0, 0};
        i32x8 gz8 = {0, 0, 0, 0, 0, 0, 0, 0};
        // 6-arg form: (g0, g1, g2, g3, extra, cpol)
        __builtin_amdgcn_tensor_load_to_lds(g0, g1, gz, gz, gz8, 0);
        __builtin_amdgcn_s_wait_tensorcnt(0);
    }
#else
    for (int j = tid; j < 16 * 128; j += 192) {
        int mm = j >> 7, k = j & 127;
        lds[mm * C1_LDKP + k] = h0[j];
    }
#endif
    __syncthreads();

    v8f c = {};
#pragma unroll
    for (int ks = 0; ks < 4; ++ks) {
        v16h a  = load_A_rowmajor(lds, C1_LDKP, ks * 32, lane);
        v16h bb = *(const v16h*)(wp + (((wave * 4 + ks) * 32 + lane) << 4));
        c = wmma_f16(a, bb, c);
    }

    const int n  = wave * 16 + (lane & 15);
    const int mb = (lane & 16) ? 8 : 0;
    if (n < FC1_N) {
#pragma unroll
        for (int r = 0; r < 8; ++r)
            h1[(r + mb) * FC1_N + n] = sigmoidf_(c[r] + bias[n]);
    }
}

// ---------------------------------------------------------------------------
// fc2 + per-batch params: out(16x10), then g1/g2/gain per batch.
// ---------------------------------------------------------------------------
__global__ void fc2_params_kernel(const float* __restrict__ h1, const float* __restrict__ w,
                                  const float* __restrict__ bias,
                                  const float* __restrict__ gb1, const float* __restrict__ gb2,
                                  const float* __restrict__ base1, float* __restrict__ params) {
    __shared__ float outsm[160];
    const int tid = threadIdx.x;
    if (tid < 160) {
        int b = tid / 10, j = tid - b * 10;
        float s = bias[j];
        for (int k = 0; k < FC1_N; ++k) s += h1[b * FC1_N + k] * w[j * FC1_N + k];
        outsm[tid] = s;
    }
    __syncthreads();
    if (tid < 16) {
        float g1 = outsm[tid * 10 + 0] + gb1[0];
        float g2 = outsm[tid * 10 + 1] + gb2[0];
        float gain = 0.0f;
        for (int j = 2; j < 10; ++j) gain += outsm[tid * 10 + j] + base1[j - 2];
        params[tid * 3 + 0] = g1;
        params[tid * 3 + 1] = g2;
        params[tid * 3 + 2] = gain;
    }
}

// ---------------------------------------------------------------------------
// Final epilogue: out = clip(img*gain, EPS, 1) ^ (mask ? g2 : g1)
// ---------------------------------------------------------------------------
__global__ void final_kernel(const float* __restrict__ img, const unsigned char* __restrict__ mask,
                             const float* __restrict__ params, float* __restrict__ out) {
    int idx = blockIdx.x * blockDim.x + threadIdx.x;   // 16*3*PLANE
    if (idx >= BATCH * 3 * PLANE) return;
    int pix = idx & (PLANE - 1);
    int t   = idx >> 18;                // b*3 + c
    int b   = t / 3;
    float gain = params[b * 3 + 2];
    unsigned char mk = mask[(b << 18) | pix];
    float g = mk ? params[b * 3 + 1] : params[b * 3 + 0];
    float v = img[idx] * gain;
    v = fminf(fmaxf(v, 1e-8f), 1.0f);
    out[idx] = __expf(g * __logf(v));
}

// ---------------------------------------------------------------------------
// Launcher
// ---------------------------------------------------------------------------
extern "C" void kernel_launch(void* const* d_in, const int* in_sizes, int n_in,
                              void* d_out, int out_size, void* d_ws, size_t ws_size,
                              hipStream_t stream) {
    const float* img   = (const float*)d_in[0];
    const float* c1w   = (const float*)d_in[1];
    const float* c1b   = (const float*)d_in[2];
    const float* c2w   = (const float*)d_in[3];
    const float* c2b   = (const float*)d_in[4];
    const float* f0w   = (const float*)d_in[5];
    const float* f0b   = (const float*)d_in[6];
    const float* f1w   = (const float*)d_in[7];
    const float* f1b   = (const float*)d_in[8];
    const float* f2w   = (const float*)d_in[9];
    const float* f2b   = (const float*)d_in[10];
    const float* gb1   = (const float*)d_in[11];
    const float* gb2   = (const float*)d_in[12];
    const float* base1 = (const float*)d_in[13];

    char* ws = (char*)d_ws;
    _Float16*      x4       = (_Float16*)(ws + OFF_BUFA);
    _Float16*      pooled1  = (_Float16*)(ws + OFF_BUFA);   // reuse after conv1 consumes x4
    _Float16*      pooled2  = (_Float16*)(ws + OFF_BUFA);   // reuse after conv2 consumes pooled1
    _Float16*      conv1out = (_Float16*)(ws + OFF_BUFB);
    _Float16*      conv2out = (_Float16*)(ws + OFF_BUFB);   // reuse after pool1 consumes conv1out
    unsigned char* maskbuf  = (unsigned char*)(ws + OFF_MASK);
    _Float16*      wp1      = (_Float16*)(ws + OFF_WP1);
    _Float16*      wp2      = (_Float16*)(ws + OFF_WP2);
    _Float16*      wpfc1    = (_Float16*)(ws + OFF_WPFC1);
    float*         acc      = (float*)(ws + OFF_ACC);
    _Float16*      h0       = (_Float16*)(ws + OFF_H0);
    float*         h1       = (float*)(ws + OFF_H1);
    float*         params   = (float*)(ws + OFF_PARAMS);

    // Pack GEMM B operands into WMMA register layout (f16)
    pack_B_kernel<<<8,  256, 0, stream>>>(c1w, wp1,   C1_OUT, 100, 4, 1);
    pack_B_kernel<<<10, 256, 0, stream>>>(c2w, wp2,   C2_OUT, 150, 5, 1);
    pack_B_kernel<<<48, 256, 0, stream>>>(f1w, wpfc1, FC1_N,  120, 4, 6);

    // Front end
    prep_kernel<<<16384, 256, 0, stream>>>(img, x4, maskbuf);
    conv1_kernel<<<32258, 256, 0, stream>>>(x4, wp1, c1b, conv1out);
    pool_kernel<<<24194, 256, 0, stream>>>(conv1out, pooled1, C1_OUT, C1_H, P1_H,
                                           BATCH * C1_OUT * P1_H * P1_H);
    conv2_kernel<<<7814, 256, 0, stream>>>(pooled1, wp2, c2b, conv2out);
    pool_kernel<<<15625, 256, 0, stream>>>(conv2out, pooled2, C2_OUT, C2_H, P2_H,
                                           BATCH * C2_OUT * P2_H * P2_H);

    // Classifier head
    zero_kernel<<<8, 256, 0, stream>>>(acc, 16 * 128);
    fc0_kernel<<<256, 32, 0, stream>>>(pooled2, f0w, acc);
    fc0_post_kernel<<<8, 256, 0, stream>>>(acc, f0b, h0);
    fc1_kernel<<<1, 192, 0, stream>>>(h0, wpfc1, f1b, h1);
    fc2_params_kernel<<<1, 256, 0, stream>>>(h1, f2w, f2b, gb1, gb2, base1, params);

    // Elementwise gamma/gain epilogue
    final_kernel<<<49152, 256, 0, stream>>>(img, maskbuf, params, (float*)d_out);
}
